// YOLOWithNMS_11836929867798
// MI455X (gfx1250) — compile-verified, matrix operand order checked by reading, not run on hardware
//
#include <hip/hip_runtime.h>

// ---------------- constants ----------------
#define N_ELEMS   4194304
#define K_TOP     2048
#define CONF_THF  0.8f
#define IOU_THF   0.6f
#define YOLO_SZF  640.0f
#define EPSF      1e-9f
#define NBINS     8192
#define CAP       8192
#define BIN_SCALE ((float)NBINS / (1.0f - CONF_THF))

typedef unsigned int u32;
typedef unsigned long long u64;

typedef __attribute__((ext_vector_type(16))) _Float16 v16h;
typedef __attribute__((ext_vector_type(8)))  float    v8f;
typedef __attribute__((ext_vector_type(4)))  unsigned int v4u;
typedef __attribute__((ext_vector_type(8)))  int      v8i;
typedef __attribute__((ext_vector_type(4)))  int      v4i;

// workspace word counts
#define W_HIST  NBINS
#define W_CTRL  8
#define W_SELS  CAP
#define W_SELI  CAP
#define W_TKS   K_TOP
#define W_TKI   K_TOP
#define W_BOX   (K_TOP * 4)
#define W_AREA  K_TOP
#define W_KEEP  64
#define W_SUP   (K_TOP * 64)
#define TOTAL_WORDS (W_HIST + W_CTRL + W_SELS + W_SELI + W_TKS + W_TKI + W_BOX + W_AREA + W_KEEP + W_SUP)

// ---------------- kernels ----------------

__global__ void init_ws_kernel(u32* ws, int nwords) {
    for (int i = blockIdx.x * blockDim.x + threadIdx.x; i < nwords;
         i += gridDim.x * blockDim.x)
        ws[i] = 0u;
}

// Pass 1: LDS histogram of conf in [0.8, 1.0]; streams conf channel as float4.
__global__ void hist_kernel(const float* __restrict__ conf, u32* __restrict__ hist) {
    __shared__ u32 sh[NBINS];
    for (int b = threadIdx.x; b < NBINS; b += blockDim.x) sh[b] = 0u;
    __syncthreads();
    const float4* c4 = (const float4*)conf;
    const int total = N_ELEMS / 4;
    for (int i = blockIdx.x * blockDim.x + threadIdx.x; i < total;
         i += gridDim.x * blockDim.x) {
        float4 v = c4[i];
        float vv[4] = {v.x, v.y, v.z, v.w};
#pragma unroll
        for (int q = 0; q < 4; ++q) {
            float cc = vv[q];
            if (cc >= CONF_THF) {
                int b = (int)((cc - CONF_THF) * BIN_SCALE);
                if (b > NBINS - 1) b = NBINS - 1;
                atomicAdd(&sh[b], 1u);
            }
        }
    }
    __syncthreads();
    for (int b = threadIdx.x; b < NBINS; b += blockDim.x)
        if (sh[b]) atomicAdd(&hist[b], sh[b]);
}

// Pass 2: find highest bin such that count(>= bin) >= K.
__global__ void cut_kernel(const u32* __restrict__ hist, u32* __restrict__ ctrl) {
    if (threadIdx.x != 0 || blockIdx.x != 0) return;
    u32 cum = 0; int cut = 0;
    for (int b = NBINS - 1; b >= 0; --b) {
        cum += hist[b];
        if (cum >= (u32)K_TOP) { cut = b; break; }
    }
    ctrl[1] = (u32)cut;
}

// Pass 3: compact every candidate at/above the cut bin.
__global__ void compact_kernel(const float* __restrict__ conf, u32* __restrict__ ctrl,
                               float* __restrict__ selScore, u32* __restrict__ selIdx) {
    const u32 cut = ctrl[1];
    for (u32 i = blockIdx.x * blockDim.x + threadIdx.x; i < (u32)N_ELEMS;
         i += gridDim.x * blockDim.x) {
        float cc = conf[i];
        if (cc >= CONF_THF) {
            int b = (int)((cc - CONF_THF) * BIN_SCALE);
            if (b > NBINS - 1) b = NBINS - 1;
            if ((u32)b >= cut) {
                u32 pos = atomicAdd(&ctrl[0], 1u);
                if (pos < (u32)CAP) { selScore[pos] = cc; selIdx[pos] = i; }
            }
        }
    }
}

// Pass 4: bitonic sort (descending) of packed keys (scoreBits<<32 | ~idx);
// tie-break matches lax.top_k (smaller index first). First K = top-k.
__global__ __launch_bounds__(1024)
void sort_topk_kernel(const float* __restrict__ selScore, const u32* __restrict__ selIdx,
                      const u32* __restrict__ ctrl,
                      float* __restrict__ tkScore, u32* __restrict__ tkIdx) {
    __shared__ u64 keys[CAP];
    const int t = threadIdx.x;
    u32 cnt = ctrl[0]; if (cnt > (u32)CAP) cnt = CAP;
    for (int i = t; i < CAP; i += 1024) {
        u64 kk = 0ull;
        if (i < (int)cnt) {
            u32 sb = __float_as_uint(selScore[i]);   // scores > 0 -> monotonic bits
            kk = ((u64)sb << 32) | (u32)(~selIdx[i]);
        }
        keys[i] = kk;
    }
    __syncthreads();
    for (int k = 2; k <= CAP; k <<= 1) {
        for (int j = k >> 1; j > 0; j >>= 1) {
            for (int i = t; i < CAP; i += 1024) {
                int ixj = i ^ j;
                if (ixj > i) {
                    bool desc = (i & k) == 0;
                    u64 A = keys[i], B = keys[ixj];
                    if (desc ? (A < B) : (A > B)) { keys[i] = B; keys[ixj] = A; }
                }
            }
            __syncthreads();
        }
    }
    for (int s = t; s < K_TOP; s += 1024) {
        u64 kk = keys[s];
        tkScore[s] = __uint_as_float((u32)(kk >> 32));
        tkIdx[s]   = ~((u32)kk);
    }
}

// Pass 5: gather cx,cy,w,h for top-k, decode boxes, areas, valid bitmask.
__global__ void gather_kernel(const float* __restrict__ pred, const float* __restrict__ img,
                              const float* __restrict__ tkScore, const u32* __restrict__ tkIdx,
                              float4* __restrict__ boxes, float* __restrict__ area,
                              u32* __restrict__ keepw) {
    int s = blockIdx.x * blockDim.x + threadIdx.x;
    if (s >= K_TOP) return;
    float sc = tkScore[s];
    bool valid = sc >= CONF_THF;
    u32 idx = valid ? tkIdx[s] : 0u;
    float cx = pred[idx];
    float cy = pred[(size_t)N_ELEMS + idx];
    float w  = pred[2ull * N_ELEMS + idx];
    float h  = pred[3ull * N_ELEMS + idx];
    float sx = img[0] * (1.0f / YOLO_SZF);
    float sy = img[1] * (1.0f / YOLO_SZF);
    float x1 = (cx - w * 0.5f) * sx, y1 = (cy - h * 0.5f) * sy;
    float x2 = (cx + w * 0.5f) * sx, y2 = (cy + h * 0.5f) * sy;
    boxes[s] = make_float4(x1, y1, x2, y2);
    area[s]  = (x2 - x1) * (y2 - y1);
    if (valid) atomicOr(&keepw[s >> 5], 1u << (s & 31));
}

// Pass 6: suppression bitmatrix. One wave32 per 16x16 tile (jt >= it only).
// - row boxes staged into LDS via Tensor Data Mover (async DMA, TENSORcnt)
// - area_i + area_j tile computed with V_WMMA_F32_16X16X32_F16
//   (A row m = [area_m, 1, 0...], B col n = [1, area_n, 0...]^T)
// - IoU threshold evaluated divide-free: inter/den > TH  <=>  den>0 && inter > TH*den
// - bits packed via wave ballot, OR'd into 2048x64-word bitmatrix.
__global__ void supmat_kernel(const float4* __restrict__ boxes,
                              const float* __restrict__ area,
                              u32* __restrict__ sup) {
    const int jt = blockIdx.x, it = blockIdx.y;
    if (jt < it) return;
    const int lane = threadIdx.x;
    const int i0 = it * 16, j0 = jt * 16;

    __shared__ float rb[64];  // 16 row boxes (x1,y1,x2,y2)

#if __has_builtin(__builtin_amdgcn_tensor_load_to_lds)
    {
        // D# group0: count=1, lds_addr, global_addr, type=2 (bits 127:126)
        u64 ga = (u64)(uintptr_t)(const void*)(boxes + i0);
        u32 la = (u32)(uintptr_t)(void*)&rb[0];
        v4u g0 = { 1u, la, (u32)(ga & 0xFFFFFFFFull),
                   (u32)((ga >> 32) & 0x1FFFFFFull) | 0x80000000u };
        // group1: data_size=4B(code 2); tensor_dim0=64, tensor_dim1=1,
        // tile_dim0=64 (one 256B row of 16 float4 boxes), stride0=64.
        v8i g1 = { (int)(2u << 16),   // wg_mask=0 | data_size=2
                   (int)(64u << 16),  // tensor_dim0 lo16 in [31:16]
                   (int)(1u << 16),   // tensor_dim0 hi=0 | tensor_dim1 lo16=1
                   (int)(64u << 16),  // tensor_dim1 hi=0 | tile_dim0=64
                   0,                 // tile_dim1=0, tile_dim2=0
                   64,                // tensor_dim0_stride lo32
                   0, 0 };
        v4i g2 = {0, 0, 0, 0};
        v4i g3 = {0, 0, 0, 0};
        v8i g4 = {0, 0, 0, 0, 0, 0, 0, 0};
        __builtin_amdgcn_tensor_load_to_lds(g0, g1, g2, g3, g4, 0);
        __builtin_amdgcn_s_wait_tensorcnt(0);
    }
#else
    if (lane < 16) {
        float4 b4 = boxes[i0 + lane];
        rb[lane * 4 + 0] = b4.x; rb[lane * 4 + 1] = b4.y;
        rb[lane * 4 + 2] = b4.z; rb[lane * 4 + 3] = b4.w;
    }
#endif
    __syncthreads();

    // WMMA area-sum tile: S[m][n] = area_i0+m + area_j0+n
    float areaA = (lane < 16) ? area[i0 + lane] : 0.0f;
    float areaB = (lane < 16) ? area[j0 + lane] : 0.0f;
    v16h a = {};
    v16h b = {};
    if (lane < 16) {
        a[0] = (_Float16)areaA; a[1] = (_Float16)1.0f;   // A[m][0]=area, A[m][1]=1
        b[0] = (_Float16)1.0f;  b[1] = (_Float16)areaB;  // B[0][n]=1,  B[1][n]=area
    }
    v8f c = {};
    c = __builtin_amdgcn_wmma_f32_16x16x32_f16(false, a, false, b,
                                               (short)0, c, false, false);

    const int n = lane & 15;
    float4 cb = boxes[j0 + n];

#pragma unroll
    for (int v = 0; v < 8; ++v) {
        int m = (lane < 16) ? v : v + 8;   // C/D layout: lanes 16-31 hold M=v+8
        int i = i0 + m, j = j0 + n;
        float lx = fmaxf(rb[4 * m + 0], cb.x);
        float ly = fmaxf(rb[4 * m + 1], cb.y);
        float rx = fminf(rb[4 * m + 2], cb.z);
        float ry = fminf(rb[4 * m + 3], cb.w);
        float wv = fmaxf(rx - lx, 0.0f), hv = fmaxf(ry - ly, 0.0f);
        float inter = wv * hv;
        float den = c[v] - inter + EPSF;   // union area + eps
        // sign-exact divide-free threshold: inter/den > TH
        bool bit = (den > 0.0f) && (inter > IOU_THF * den) && (j > i);
        u64 bal = __ballot(bit);
        u32 lo = (u32)(bal & 0xFFFFu);          // rows m=v   (lanes 0-15)
        u32 hi = (u32)((bal >> 16) & 0xFFFFu);  // rows m=v+8 (lanes 16-31)
        if (lane == 0) {
            int word = j0 >> 5, sh = j0 & 31;
            atomicOr(&sup[(u32)(i0 + v) * 64u + word], lo << sh);
            atomicOr(&sup[(u32)(i0 + v + 8) * 64u + word], hi << sh);
        }
    }
}

// Pass 7: greedy NMS scan. Single wave32; 2048-bit keep mask lives in
// 2 VGPRs per lane; pivot bit broadcast via shuffle, no barriers.
__global__ void nms_scan_kernel(const u32* __restrict__ sup, u32* __restrict__ keepw) {
    const int lane = threadIdx.x;
    u32 kw0 = keepw[2 * lane];
    u32 kw1 = keepw[2 * lane + 1];
    for (int i = 0; i < K_TOP; ++i) {
        int w = i >> 5;
        int owner = w >> 1;
        u32 src = (w & 1) ? kw1 : kw0;
        u32 word = __shfl(src, owner, 32);
        if ((word >> (i & 31)) & 1u) {
            kw0 &= ~sup[i * 64 + 2 * lane];
            kw1 &= ~sup[i * 64 + 2 * lane + 1];
        }
    }
    keepw[2 * lane] = kw0;
    keepw[2 * lane + 1] = kw1;
}

// Pass 8: masked outputs: crops (K*4), boxes (K*4), scores (K).
__global__ void output_kernel(const float4* __restrict__ boxes,
                              const float* __restrict__ tkScore,
                              const u32* __restrict__ keepw,
                              const float* __restrict__ img,
                              float* __restrict__ out) {
    int s = blockIdx.x * blockDim.x + threadIdx.x;
    if (s >= K_TOP) return;
    bool keep = (keepw[s >> 5] >> (s & 31)) & 1u;
    float4 b = boxes[s];
    float W = img[0], H = img[1];
    float ccx = (b.x + b.z) * 0.5f, ccy = (b.y + b.w) * 0.5f;
    float rect = fmaxf(b.z - b.x, b.w - b.y);
    float crop = fminf(fminf(W, H), rect * 3.0f);
    float x1 = ccx - crop * 0.5f, x2 = ccx + crop * 0.5f;
    float y1 = ccy - crop * 0.5f, y2 = ccy + crop * 0.5f;
    float xs = fmaxf(-x1, 0.0f) - fmaxf(x2 - W, 0.0f);
    float ys = fmaxf(-y1, 0.0f) - fmaxf(y2 - H, 0.0f);
    float cx1 = fminf(fmaxf(x1 + xs, 0.0f), W);
    float cy1 = fminf(fmaxf(y1 + ys, 0.0f), H);
    float cx2 = fminf(fmaxf(x2 + xs, 0.0f), W);
    float cy2 = fminf(fmaxf(y2 + ys, 0.0f), H);

    out[4 * s + 0] = keep ? cx1 : 0.0f;
    out[4 * s + 1] = keep ? cy1 : 0.0f;
    out[4 * s + 2] = keep ? cx2 : 0.0f;
    out[4 * s + 3] = keep ? cy2 : 0.0f;
    out[4 * K_TOP + 4 * s + 0] = keep ? b.x : 0.0f;
    out[4 * K_TOP + 4 * s + 1] = keep ? b.y : 0.0f;
    out[4 * K_TOP + 4 * s + 2] = keep ? b.z : 0.0f;
    out[4 * K_TOP + 4 * s + 3] = keep ? b.w : 0.0f;
    out[8 * K_TOP + s] = keep ? tkScore[s] : 0.0f;
}

// ---------------- host launcher ----------------
extern "C" void kernel_launch(void* const* d_in, const int* in_sizes, int n_in,
                              void* d_out, int out_size, void* d_ws, size_t ws_size,
                              hipStream_t stream) {
    (void)in_sizes; (void)n_in; (void)out_size; (void)ws_size;
    const float* pred = (const float*)d_in[0];
    const float* img  = (const float*)d_in[1];
    const float* conf = pred + 4ull * N_ELEMS;

    u32*   ws       = (u32*)d_ws;
    u32*   hist     = ws;
    u32*   ctrl     = hist + W_HIST;
    float* selScore = (float*)(ctrl + W_CTRL);
    u32*   selIdx   = (u32*)(selScore + W_SELS);
    float* tkScore  = (float*)(selIdx + W_SELI);
    u32*   tkIdx    = (u32*)(tkScore + W_TKS);
    float4* boxes   = (float4*)(tkIdx + W_TKI);
    float* area     = (float*)((u32*)boxes + W_BOX);
    u32*   keepw    = (u32*)(area + W_AREA);
    u32*   sup      = keepw + W_KEEP;

    init_ws_kernel<<<256, 256, 0, stream>>>(ws, TOTAL_WORDS);
    hist_kernel<<<1024, 256, 0, stream>>>(conf, hist);
    cut_kernel<<<1, 1, 0, stream>>>(hist, ctrl);
    compact_kernel<<<2048, 256, 0, stream>>>(conf, ctrl, selScore, selIdx);
    sort_topk_kernel<<<1, 1024, 0, stream>>>(selScore, selIdx, ctrl, tkScore, tkIdx);
    gather_kernel<<<(K_TOP + 255) / 256, 256, 0, stream>>>(pred, img, tkScore, tkIdx,
                                                           boxes, area, keepw);
    supmat_kernel<<<dim3(128, 128), 32, 0, stream>>>(boxes, area, sup);
    nms_scan_kernel<<<1, 32, 0, stream>>>(sup, keepw);
    output_kernel<<<(K_TOP + 255) / 256, 256, 0, stream>>>(boxes, tkScore, keepw, img,
                                                           (float*)d_out);
}
